// TSModel_584115552420
// MI455X (gfx1250) — matrix operand order repaired and checked by reading, not compiled, not run on hardware
//
#include <hip/hip_runtime.h>
#include <hip/hip_bf16.h>
#include <math.h>

// ---------------------------------------------------------------------------
// Types for CDNA5 WMMA
// ---------------------------------------------------------------------------
typedef __attribute__((ext_vector_type(16))) __bf16 v16bf;
typedef __attribute__((ext_vector_type(2)))  __bf16 v2bf;
typedef __attribute__((ext_vector_type(8)))  float  v8f;

// build a packed bf16 pair; ISel fuses the two truncs into v_cvt_pk_bf16_f32
static __device__ __forceinline__ v2bf pk2(float a, float b)
{
    v2bf r;
    r[0] = (__bf16)a;
    r[1] = (__bf16)b;
    return r;
}

// ---------------------------------------------------------------------------
// Generic batched GEMM:  C[z] = alpha * A[z] x B[z] + bias  (opt. ReLU)
//   A: f32 row-major [M x K], lda          (M % 128 == 0, K % 32 == 0 assumed)
//   B: f32, transB==0 -> [K x N] ldb ; transB==1 -> [N x K] ldb
//   C: f32 row-major [M x N], ldc          (N may be < BN; stores are guarded,
//                                           B over-reads stay inside workspace)
// z decomposition: z1 = z / zdiv, z2 = z % zdiv; offsets z1*s?1 + z2*s?2
// Block tile 128x128, 8 wave32 (4x2), wave tile 32x64 = 2x4 WMMA 16x16x32 bf16.
// Double-buffered LDS + register prefetch: one barrier per K-step, global
// float4 loads and LDS tile stores overlap the 8 WMMAs.
// ---------------------------------------------------------------------------
#define BM 128
#define BN 128
#define KT 32
#define LDK (KT + 2)

__global__ __launch_bounds__(256)
void gemm_bf16_kernel(const float* __restrict__ A, const float* __restrict__ B,
                      const float* __restrict__ bias, float* __restrict__ C,
                      int M, int N, int K, int lda, int ldb, int ldc,
                      int transB, int relu, float alpha, int zdiv,
                      long long sA1, long long sA2,
                      long long sB1, long long sB2,
                      long long sC1, long long sC2)
{
    __shared__ __bf16 sA[2][BM][LDK];   // [buf][m][k], k-contiguous
    __shared__ __bf16 sB[2][BN][LDK];   // [buf][n][k], k-contiguous

    const int z  = blockIdx.z;
    const int z1 = z / zdiv, z2 = z % zdiv;
    A += z1 * sA1 + z2 * sA2;
    B += z1 * sB1 + z2 * sB2;
    C += z1 * sC1 + z2 * sC2;

    const int t     = threadIdx.x;
    const int lane  = t & 31;
    const int w     = t >> 5;
    const int wm    = (w >> 1) * 32;   // 4 waves in M
    const int wn    = (w & 1) * 64;    // 2 waves in N
    const int m0    = blockIdx.x * BM;
    const int n0    = blockIdx.y * BN;
    const int lhalf = lane >> 4;
    const int lrow  = lane & 15;
    const int kbA   = lhalf * 8;       // A frag K base (ISA 7.12.2)
    const int kbB   = lhalf * 16;      // unused; B frag uses A's k order (kb = ka)

    // staging coordinates
    const int am  = t >> 1;            // 0..127  (A row)
    const int ak  = (t & 1) * 16;      // 0/16    (A k base, 16 consecutive)
    const int bn  = t >> 1;            // 0..127  (B n-row, transB path)
    const int bk  = (t & 1) * 16;
    const int ck2 = (t >> 4) * 2;      // 0..30   (B k-pair, non-trans path)
    const int cn8 = (t & 15) * 8;      // 8 consecutive n

    float4 ra[4], rb[4];

    auto load_tiles = [&](int k0) {
        const float4* ap = (const float4*)(A + (size_t)(m0 + am) * lda + k0 + ak);
#pragma unroll
        for (int j = 0; j < 4; ++j) ra[j] = ap[j];
        if (transB) {
            const float4* bp = (const float4*)(B + (size_t)(n0 + bn) * ldb + k0 + bk);
#pragma unroll
            for (int j = 0; j < 4; ++j) rb[j] = bp[j];
        } else {
            const float4* b0 = (const float4*)(B + (size_t)(k0 + ck2) * ldb + n0 + cn8);
            const float4* b1 = (const float4*)(B + (size_t)(k0 + ck2 + 1) * ldb + n0 + cn8);
            rb[0] = b0[0]; rb[1] = b0[1];
            rb[2] = b1[0]; rb[3] = b1[1];
        }
    };

    auto store_tiles = [&](int p) {
        v2bf* da = (v2bf*)&sA[p][am][ak];
#pragma unroll
        for (int j = 0; j < 4; ++j) {
            da[2 * j]     = pk2(ra[j].x, ra[j].y);
            da[2 * j + 1] = pk2(ra[j].z, ra[j].w);
        }
        if (transB) {
            v2bf* db = (v2bf*)&sB[p][bn][bk];
#pragma unroll
            for (int j = 0; j < 4; ++j) {
                db[2 * j]     = pk2(rb[j].x, rb[j].y);
                db[2 * j + 1] = pk2(rb[j].z, rb[j].w);
            }
        } else {
            // rb[0..1] = row k (8 n), rb[2..3] = row k+1 (8 n); pack along k
            *(v2bf*)&sB[p][cn8 + 0][ck2] = pk2(rb[0].x, rb[2].x);
            *(v2bf*)&sB[p][cn8 + 1][ck2] = pk2(rb[0].y, rb[2].y);
            *(v2bf*)&sB[p][cn8 + 2][ck2] = pk2(rb[0].z, rb[2].z);
            *(v2bf*)&sB[p][cn8 + 3][ck2] = pk2(rb[0].w, rb[2].w);
            *(v2bf*)&sB[p][cn8 + 4][ck2] = pk2(rb[1].x, rb[3].x);
            *(v2bf*)&sB[p][cn8 + 5][ck2] = pk2(rb[1].y, rb[3].y);
            *(v2bf*)&sB[p][cn8 + 6][ck2] = pk2(rb[1].z, rb[3].z);
            *(v2bf*)&sB[p][cn8 + 7][ck2] = pk2(rb[1].w, rb[3].w);
        }
    };

    v8f acc[2][4] = {};

    load_tiles(0);
    store_tiles(0);
    __syncthreads();

    int p = 0;
    for (int k0 = 0; k0 < K; k0 += KT) {
        const bool more = (k0 + KT) < K;
        if (more) load_tiles(k0 + KT);          // global prefetch, in flight below

        const __bf16 (*cA)[LDK] = sA[p];
        const __bf16 (*cB)[LDK] = sB[p];
        v16bf af[2], bf[4];
#pragma unroll
        for (int e = 0; e < 16; ++e) {
            const int ka = kbA + ((e < 8) ? e : (8 + e));
            const int kb = ka;
#pragma unroll
            for (int i = 0; i < 2; ++i) af[i][e] = cA[wm + i * 16 + lrow][ka];
#pragma unroll
            for (int j = 0; j < 4; ++j) bf[j][e] = cB[wn + j * 16 + lrow][kb];
        }
#pragma unroll
        for (int i = 0; i < 2; ++i)
#pragma unroll
            for (int j = 0; j < 4; ++j)
                acc[i][j] = __builtin_amdgcn_wmma_f32_16x16x32_bf16(
                    false, af[i], false, bf[j], (short)0, acc[i][j], false, false);

        if (more) store_tiles(p ^ 1);           // fill other buffer behind WMMAs
        __syncthreads();
        p ^= 1;
    }

    // ---- epilogue: C/D layout n = lane&15 ; m = vgpr + 8*lanehalf
#pragma unroll
    for (int i = 0; i < 2; ++i) {
#pragma unroll
        for (int j = 0; j < 4; ++j) {
            const int nl = n0 + wn + j * 16 + lrow;
            if (nl >= N) continue;
            const float bs = bias ? bias[nl] : 0.0f;
#pragma unroll
            for (int v = 0; v < 8; ++v) {
                const int ml = m0 + wm + i * 16 + lhalf * 8 + v;
                float val = alpha * acc[i][j][v] + bs;
                if (relu) val = fmaxf(val, 0.0f);
                C[(size_t)ml * ldc + nl] = val;
            }
        }
    }
}

// ---------------------------------------------------------------------------
// Embedding: out[r, d<F] = in[r,d] ; out[r, d>=F] = emb_b + emb_w . in_row
// then + pos_table[1 + (r % S)]
// ---------------------------------------------------------------------------
__global__ void embed_kernel(const float* __restrict__ in, const float* __restrict__ ew,
                             const float* __restrict__ eb, const float* __restrict__ pos,
                             float* __restrict__ out, int S, int F, int Dm)
{
    const int idx = blockIdx.x * blockDim.x + threadIdx.x;
    const int d = idx % Dm;
    const int r = idx / Dm;
    const int s = r % S;
    const float* xr = in + (size_t)r * F;
    float v;
    if (d < F) {
        v = xr[d];
    } else {
        const int o = d - F;
        const float* wr = ew + (size_t)o * F;
        float a = eb[o];
        for (int f = 0; f < F; ++f) a += xr[f] * wr[f];
        v = a;
    }
    out[idx] = v + pos[(size_t)(s + 1) * Dm + d];
}

// ---------------------------------------------------------------------------
// Pad flags: flag[b*S+s] = (sum_f in[b,s,f]) <= -9999
// ---------------------------------------------------------------------------
__global__ void pad_kernel(const float* __restrict__ in, int* __restrict__ flags, int F)
{
    const int r = blockIdx.x * blockDim.x + threadIdx.x;
    const float* xr = in + (size_t)r * F;
    float s = 0.0f;
    for (int f = 0; f < F; ++f) s += xr[f];
    flags[r] = (s <= -9999.0f) ? 1 : 0;
}

// ---------------------------------------------------------------------------
// Mask + softmax over rows of scores [B,H,Sq,Sk]; causal adds k > q mask.
// ---------------------------------------------------------------------------
__global__ __launch_bounds__(256)
void softmax_kernel(float* __restrict__ sc, const int* __restrict__ pad,
                    int Sq, int Sk, int H, int causal)
{
    const int r  = blockIdx.x;
    const int qi = r % Sq;
    const int b  = r / (H * Sq);
    float* row = sc + (size_t)r * Sk;
    const int* pr = pad + (size_t)b * Sk;
    const int t = threadIdx.x;
    __shared__ float red[256];

    float lmax = -3.0e38f;
    for (int k = t; k < Sk; k += 256) {
        float x = row[k];
        if (pr[k] || (causal && k > qi)) x = -1e9f;
        row[k] = x;
        lmax = fmaxf(lmax, x);
    }
    red[t] = lmax; __syncthreads();
    for (int o = 128; o > 0; o >>= 1) { if (t < o) red[t] = fmaxf(red[t], red[t + o]); __syncthreads(); }
    const float m = red[0]; __syncthreads();

    float ls = 0.0f;
    for (int k = t; k < Sk; k += 256) { float e = __expf(row[k] - m); row[k] = e; ls += e; }
    red[t] = ls; __syncthreads();
    for (int o = 128; o > 0; o >>= 1) { if (t < o) red[t] += red[t + o]; __syncthreads(); }
    const float inv = 1.0f / red[0];
    for (int k = t; k < Sk; k += 256) row[k] *= inv;
}

// ---------------------------------------------------------------------------
// Residual + LayerNorm (unbiased std, / (std + eps)) per row of D=512.
// out = g * (x - mean) / (std + 1e-6) + b ; x = resid + delta
// ---------------------------------------------------------------------------
__global__ __launch_bounds__(256)
void ln_kernel(const float* __restrict__ resid, const float* __restrict__ delta,
               const float* __restrict__ gamma, const float* __restrict__ beta,
               float* __restrict__ out, int D)
{
    const int r = blockIdx.x;
    const int t = threadIdx.x;
    __shared__ float buf[512];
    __shared__ float red[256];
    const float* xr = resid + (size_t)r * D;
    const float* dr = delta + (size_t)r * D;

    float ls = 0.0f;
    for (int d = t; d < D; d += 256) { float v = xr[d] + dr[d]; buf[d] = v; ls += v; }
    red[t] = ls; __syncthreads();
    for (int o = 128; o > 0; o >>= 1) { if (t < o) red[t] += red[t + o]; __syncthreads(); }
    const float mean = red[0] / D; __syncthreads();

    float ss = 0.0f;
    for (int d = t; d < D; d += 256) { float v = buf[d] - mean; ss += v * v; }
    red[t] = ss; __syncthreads();
    for (int o = 128; o > 0; o >>= 1) { if (t < o) red[t] += red[t + o]; __syncthreads(); }
    const float stdv = sqrtf(red[0] / (float)(D - 1));
    const float inv  = 1.0f / (stdv + 1e-6f);
    for (int d = t; d < D; d += 256)
        out[(size_t)r * D + d] = gamma[d] * (buf[d] - mean) * inv + beta[d];
}

// ---------------------------------------------------------------------------
// Host orchestration
// ---------------------------------------------------------------------------
static void launch_gemm(hipStream_t s, const float* A, const float* B, const float* bias,
                        float* C, int M, int N, int K, int lda, int ldb, int ldc,
                        int transB, int relu, float alpha, int Z, int zdiv,
                        long long sA1, long long sA2, long long sB1, long long sB2,
                        long long sC1, long long sC2)
{
    dim3 g((M + BM - 1) / BM, (N + BN - 1) / BN, Z);
    gemm_bf16_kernel<<<g, 256, 0, s>>>(A, B, bias, C, M, N, K, lda, ldb, ldc,
                                       transB, relu, alpha, zdiv,
                                       sA1, sA2, sB1, sB2, sC1, sC2);
}

extern "C" void kernel_launch(void* const* d_in, const int* in_sizes, int n_in,
                              void* d_out, int out_size, void* d_ws, size_t ws_size,
                              hipStream_t stream)
{
    (void)in_sizes; (void)n_in; (void)out_size; (void)ws_size;
    const int Bb = 8, S = 512, F = 32, Dm = 512, Dff = 2048, Hh = 8, DKc = 64, Lc = 4;
    const int BS = Bb * S;            // 4096 rows
    const int BH = Bb * Hh;           // 64 (b,h) pairs
    const float SCALE = 1.0f / 8.0f;  // 1/sqrt(64)

    const float* enc_in   = (const float*)d_in[0];
    const float* dec_in   = (const float*)d_in[1];
    const float* emb_w    = (const float*)d_in[2];
    const float* emb_b    = (const float*)d_in[3];
    const float* pos      = (const float*)d_in[4];
    const float* e_qkv_w  = (const float*)d_in[5];
    const float* e_qkv_b  = (const float*)d_in[6];
    const float* e_pw     = (const float*)d_in[7];
    const float* e_pb     = (const float*)d_in[8];
    const float* e_ln1    = (const float*)d_in[9];
    const float* e_w1     = (const float*)d_in[10];
    const float* e_b1     = (const float*)d_in[11];
    const float* e_w2     = (const float*)d_in[12];
    const float* e_b2     = (const float*)d_in[13];
    const float* e_ln2    = (const float*)d_in[14];
    const float* ds_qkv_w = (const float*)d_in[15];
    const float* ds_qkv_b = (const float*)d_in[16];
    const float* ds_pw    = (const float*)d_in[17];
    const float* ds_pb    = (const float*)d_in[18];
    const float* d_ln1    = (const float*)d_in[19];
    const float* dc_qkv_w = (const float*)d_in[20];
    const float* dc_qkv_b = (const float*)d_in[21];
    const float* dc_pw    = (const float*)d_in[22];
    const float* dc_pb    = (const float*)d_in[23];
    const float* d_ln2    = (const float*)d_in[24];
    const float* d_w1     = (const float*)d_in[25];
    const float* d_b1     = (const float*)d_in[26];
    const float* d_w2     = (const float*)d_in[27];
    const float* d_w2b    = (const float*)d_in[28];
    const float* d_ln3    = (const float*)d_in[29];
    float* out = (float*)d_out;

    // workspace layout (floats)
    const size_t ACT = (size_t)BS * Dm;          // 2,097,152
    float* ws   = (float*)d_ws;
    float* enc  = ws;
    float* dec  = enc  + ACT;
    float* q    = dec  + ACT;
    float* kb   = q    + ACT;
    float* vb   = kb   + ACT;
    float* ctx  = vb   + ACT;
    float* tmp  = ctx  + ACT;
    float* ffn  = tmp  + ACT;                    // BS * Dff
    float* scr  = ffn  + (size_t)BS * Dff;       // B*H*S*S
    int*   epad = (int*)(scr + (size_t)BH * S * S);
    int*   dpad = epad + BS;

    // embeddings + pos encoding
    embed_kernel<<<(BS * Dm) / 256, 256, 0, stream>>>(enc_in, emb_w, emb_b, pos, enc, S, F, Dm);
    embed_kernel<<<(BS * Dm) / 256, 256, 0, stream>>>(dec_in, emb_w, emb_b, pos, dec, S, F, Dm);
    pad_kernel<<<BS / 256, 256, 0, stream>>>(enc_in, epad, F);
    pad_kernel<<<BS / 256, 256, 0, stream>>>(dec_in, dpad, F);

    const long long SB = (long long)S * Dm;      // per-batch stride in [B,S,Dm]
    const long long SS = (long long)S * S;       // per-(b,h) scores stride

    // one attention pass: x -> q ; kvsrc -> k,v ; softmax(mask) ; ctx ; proj -> tmp
    auto attention = [&](const float* x, const float* kvsrc,
                         const float* qkvW, const float* qkvB,
                         const float* pw, const float* pb,
                         const int* pad, int causal) {
        launch_gemm(stream, x,     qkvW,                   qkvB,        q,  BS, Dm, Dm, Dm, Dm, Dm, 0, 0, 1.0f, 1, 1, 0,0,0,0,0,0);
        launch_gemm(stream, kvsrc, qkvW + (size_t)Dm*Dm,   qkvB + Dm,   kb, BS, Dm, Dm, Dm, Dm, Dm, 0, 0, 1.0f, 1, 1, 0,0,0,0,0,0);
        launch_gemm(stream, kvsrc, qkvW + (size_t)2*Dm*Dm, qkvB + 2*Dm, vb, BS, Dm, Dm, Dm, Dm, Dm, 0, 0, 1.0f, 1, 1, 0,0,0,0,0,0);
        // scores[b,h] = Q[b,:,h*64:] x K[b,:,h*64:]^T * scale
        launch_gemm(stream, q, kb, nullptr, scr, S, S, DKc, Dm, Dm, S, 1, 0, SCALE,
                    BH, Hh, SB, DKc, SB, DKc, (long long)Hh * SS, SS);
        softmax_kernel<<<BH * S, 256, 0, stream>>>(scr, pad, S, S, Hh, causal);
        // ctx[b,:,h*64:] = P[b,h] x V[b,:,h*64:]
        launch_gemm(stream, scr, vb, nullptr, ctx, S, DKc, S, S, Dm, Dm, 0, 0, 1.0f,
                    BH, Hh, (long long)Hh * SS, SS, SB, DKc, SB, DKc);
        launch_gemm(stream, ctx, pw, pb, tmp, BS, Dm, Dm, Dm, Dm, Dm, 0, 0, 1.0f, 1, 1, 0,0,0,0,0,0);
    };

    auto ffn_block = [&](const float* x, const float* w1, const float* b1,
                         const float* w2, const float* b2) {
        launch_gemm(stream, x,   w1, b1, ffn, BS, Dff, Dm, Dm, Dm,  Dff, 1, 1, 1.0f, 1, 1, 0,0,0,0,0,0);
        launch_gemm(stream, ffn, w2, b2, tmp, BS, Dm, Dff, Dff, Dff, Dm, 1, 0, 1.0f, 1, 1, 0,0,0,0,0,0);
    };

    // ---- encoder ----
    for (int i = 0; i < Lc; ++i) {
        attention(enc, enc,
                  e_qkv_w + (size_t)i * 3 * Dm * Dm, e_qkv_b + (size_t)i * 3 * Dm,
                  e_pw + (size_t)i * Dm * Dm, e_pb + (size_t)i * Dm, epad, 0);
        ln_kernel<<<BS, 256, 0, stream>>>(enc, tmp, e_ln1 + (size_t)i * 2 * Dm,
                                          e_ln1 + (size_t)i * 2 * Dm + Dm, enc, Dm);
        ffn_block(enc, e_w1 + (size_t)i * Dff * Dm, e_b1 + (size_t)i * Dff,
                  e_w2 + (size_t)i * Dm * Dff, e_b2 + (size_t)i * Dm);
        ln_kernel<<<BS, 256, 0, stream>>>(enc, tmp, e_ln2 + (size_t)i * 2 * Dm,
                                          e_ln2 + (size_t)i * 2 * Dm + Dm, enc, Dm);
    }

    // ---- decoder ----
    for (int i = 0; i < Lc; ++i) {
        // self-attention (pad + causal)
        attention(dec, dec,
                  ds_qkv_w + (size_t)i * 3 * Dm * Dm, ds_qkv_b + (size_t)i * 3 * Dm,
                  ds_pw + (size_t)i * Dm * Dm, ds_pb + (size_t)i * Dm, dpad, 1);
        ln_kernel<<<BS, 256, 0, stream>>>(dec, tmp, d_ln1 + (size_t)i * 2 * Dm,
                                          d_ln1 + (size_t)i * 2 * Dm + Dm, dec, Dm);
        // cross-attention (enc pad)
        attention(dec, enc,
                  dc_qkv_w + (size_t)i * 3 * Dm * Dm, dc_qkv_b + (size_t)i * 3 * Dm,
                  dc_pw + (size_t)i * Dm * Dm, dc_pb + (size_t)i * Dm, epad, 0);
        ln_kernel<<<BS, 256, 0, stream>>>(dec, tmp, d_ln2 + (size_t)i * 2 * Dm,
                                          d_ln2 + (size_t)i * 2 * Dm + Dm, dec, Dm);
        // FFN
        ffn_block(dec, d_w1 + (size_t)i * Dff * Dm, d_b1 + (size_t)i * Dff,
                  d_w2 + (size_t)i * Dm * Dff, d_w2b + (size_t)i * Dm);
        float* dst = (i == Lc - 1) ? out : dec;
        ln_kernel<<<BS, 256, 0, stream>>>(dec, tmp, d_ln3 + (size_t)i * 2 * Dm,
                                          d_ln3 + (size_t)i * 2 * Dm + Dm, dst, Dm);
    }
}
